// DFlashDecoderLayer_37383395344453
// MI455X (gfx1250) — matrix operand order repaired and checked by reading, not compiled, not run on hardware
//
#include <hip/hip_runtime.h>
#include <cstddef>

// ---------------------------------------------------------------------------
// DFlash decoder layer for gfx1250 (MI455X): self-attn + cross-attn + MLP.
// All GEMMs run through v_wmma_f32_16x16x32_f16 (f16 A/B staged through LDS
// in fragment-major layouts -> ds_load_b128 only, f32 accumulate).
// K-loop is register->LDS double-buffered (1 barrier/iter) with
// global_prefetch_b8 two tiles ahead. Compute bound: ~120 GFLOP vs ~100MB.
// ---------------------------------------------------------------------------

typedef _Float16 h16;
typedef __attribute__((ext_vector_type(16))) _Float16 v16h;
typedef __attribute__((ext_vector_type(8)))  _Float16 v8h;
typedef __attribute__((ext_vector_type(8)))  float    v8f;

#define GEMM_BM 128
#define GEMM_BN 64
#define GEMM_BK 32
#define ASTR (GEMM_BK + 8)   // A tile LDS row stride (halves): 80B, 16B-aligned
#define BKSTR (GEMM_BK + 8)  // B tile LDS col stride (halves): Bs[n][k] layout

enum { EPI_NONE = 0, EPI_ADD = 1, EPI_SILU = 2, EPI_MUL = 3 };

// C[M,N] = epilogue( alpha * A[M,K] @ B[K,N] , R )
// BTRANS=1: B given as Bsrc[N,K] row-major (B[k][n] = Bsrc[n*ldb+k]) -> K^T
// operand of the attention-score GEMM.
template <int EPI, bool BTRANS>
__global__ __launch_bounds__(256) void gemm_wmma(
    const float* __restrict__ A, int lda,
    const float* __restrict__ B, int ldb,
    float* __restrict__ C, int ldc,
    const float* __restrict__ R, int ldr,
    int M, int N, int K, float alpha)
{
  __shared__ alignas(16) h16 As[2][GEMM_BM * ASTR];   // ping-pong A tiles
  __shared__ alignas(16) h16 Bs[2][GEMM_BN * BKSTR];  // ping-pong B tiles (Bs[n][k])

  const int tid  = threadIdx.x;
  const int wave = tid >> 5;
  const int lane = tid & 31;
  const int l15  = lane & 15;
  const int hi   = lane >> 4;      // lane-half select (WMMA operand layout)

  const int wm = wave & 3;         // 4 waves over M: 32-row strips
  const int wn = wave >> 2;        // 2 waves over N: 32-col strips

  const int row0 = blockIdx.y * GEMM_BM;
  const int col0 = blockIdx.x * GEMM_BN;

  // per-thread cooperative-load coordinates
  const int ar  = tid >> 1;          // A: row within tile
  const int ac  = (tid & 1) * 16;    // A: 16-half column sub-range
  const int bn  = tid & 63;          // B normal: owned column (coalesced)
  const int bkc = (tid >> 6) * 8;    // B normal: k sub-range
  const int bn2 = tid >> 2;          // B trans: owned column
  const int bc2 = (tid & 3) * 8;     // B trans: k sub-range

  float areg[16];
  float breg[8];

  // fetch tile at k-offset kk into registers (global -> VGPR)
  auto fetch = [&](int kk) {
    const float* sa = A + (size_t)(row0 + ar) * lda + (kk + ac);
#pragma unroll
    for (int j = 0; j < 16; ++j) areg[j] = sa[j];
    if (!BTRANS) {
      const float* sb = B + (size_t)(kk + bkc) * ldb + (col0 + bn);
#pragma unroll
      for (int j = 0; j < 8; ++j) breg[j] = sb[(size_t)j * ldb];
    } else {
      const float* sb = B + (size_t)(col0 + bn2) * ldb + (kk + bc2);
#pragma unroll
      for (int j = 0; j < 8; ++j) breg[j] = sb[j];
    }
  };

  // convert + stage registers into LDS buffer `buf` (VGPR -> f16 LDS, b128)
  auto stage = [&](int buf) {
    v8h lo, hiv, t;
#pragma unroll
    for (int j = 0; j < 8; ++j) lo[j]  = (h16)areg[j];
#pragma unroll
    for (int j = 0; j < 8; ++j) hiv[j] = (h16)areg[8 + j];
    *(v8h*)&As[buf][ar * ASTR + ac]     = lo;
    *(v8h*)&As[buf][ar * ASTR + ac + 8] = hiv;
#pragma unroll
    for (int j = 0; j < 8; ++j) t[j] = (h16)breg[j];
    if (!BTRANS) *(v8h*)&Bs[buf][bn * BKSTR + bkc]  = t;
    else         *(v8h*)&Bs[buf][bn2 * BKSTR + bc2] = t;
  };

  v8f acc[2][2] = {};              // 32x32 f32 macro-tile per wave

  // compute 4 WMMAs from LDS buffer `buf`
  auto compute = [&](int buf) {
    // A fragments (16-bit A 16x32 layout, ISA 7.12.2):
    // lanes 0-15: halves K{0..7} ++ K{16..23}; lanes 16-31: K{8..15} ++ K{24..31}
    v16h af[2];
#pragma unroll
    for (int i = 0; i < 2; ++i) {
      const h16* arow = &As[buf][(wm * 32 + i * 16 + l15) * ASTR];
      v8h a0 = *(const v8h*)(arow + hi * 8);
      v8h a1 = *(const v8h*)(arow + 16 + hi * 8);
      af[i] = __builtin_shufflevector(a0, a1, 0,1,2,3,4,5,6,7,8,9,10,11,12,13,14,15);
    }
    // B fragments (dense 32x16): lanes 0-15 K{0..7} ++ K{16..23}; lanes 16-31 K{8..15} ++ K{24..31}
    v16h bf[2];
#pragma unroll
    for (int j = 0; j < 2; ++j) {
      const h16* bcol = &Bs[buf][(wn * 32 + j * 16 + l15) * BKSTR];
      v8h b0 = *(const v8h*)(bcol + hi * 8);
      v8h b1 = *(const v8h*)(bcol + 16 + hi * 8);
      bf[j] = __builtin_shufflevector(b0, b1, 0,1,2,3,4,5,6,7,8,9,10,11,12,13,14,15);
    }
#pragma unroll
    for (int i = 0; i < 2; ++i)
#pragma unroll
      for (int j = 0; j < 2; ++j)
        acc[i][j] = __builtin_amdgcn_wmma_f32_16x16x32_f16(
            false, af[i], false, bf[j], (short)0, acc[i][j], false, false);
  };

  // ---- software-pipelined K loop: fetch(t+1) || compute(t), 1 barrier/iter
  const int T = K / GEMM_BK;
  fetch(0);
  stage(0);
  __syncthreads();
  for (int t = 0; t < T; ++t) {
    if (t + 1 < T) {
      fetch((t + 1) * GEMM_BK);
      if (t + 2 < T) {   // pull tile t+2 toward the caches (global_prefetch_b8)
        const int kp = (t + 2) * GEMM_BK;
        __builtin_prefetch(A + (size_t)(row0 + ar) * lda + (kp + ac), 0, 3);
        if (!BTRANS)
          __builtin_prefetch(B + (size_t)(kp + bkc) * ldb + (col0 + bn), 0, 3);
        else
          __builtin_prefetch(B + (size_t)(col0 + bn2) * ldb + (kp + bc2), 0, 3);
      }
    }
    compute(t & 1);
    if (t + 1 < T) stage((t + 1) & 1);
    __syncthreads();
  }

  // --- epilogue + writeback. C layout: VGPR g -> M = hi*8 + g, N = l15. ----
#pragma unroll
  for (int i = 0; i < 2; ++i) {
#pragma unroll
    for (int j = 0; j < 2; ++j) {
#pragma unroll
      for (int g = 0; g < 8; ++g) {
        const int m = row0 + wm * 32 + i * 16 + hi * 8 + g;
        const int n = col0 + wn * 32 + j * 16 + l15;
        float v = acc[i][j][g] * alpha;
        if (EPI == EPI_ADD)  v += R[(size_t)m * ldr + n];
        if (EPI == EPI_SILU) v  = v / (1.0f + __expf(-v));
        if (EPI == EPI_MUL)  v *= R[(size_t)m * ldr + n];
        C[(size_t)m * ldc + n] = v;
      }
    }
  }
}

// ---------------------------------------------------------------------------
__global__ __launch_bounds__(256) void rmsnorm_k(
    const float* __restrict__ x, const float* __restrict__ w,
    float* __restrict__ y, int H)
{
  __shared__ float red[8];
  __shared__ float scale;
  const int row = blockIdx.x;
  const float* xr = x + (size_t)row * H;
  float s = 0.f;
  for (int i = threadIdx.x; i < H; i += 256) { float v = xr[i]; s += v * v; }
#pragma unroll
  for (int off = 16; off; off >>= 1) s += __shfl_down(s, off, 32);
  if ((threadIdx.x & 31) == 0) red[threadIdx.x >> 5] = s;
  __syncthreads();
  if (threadIdx.x == 0) {
    float t = 0.f;
    for (int i = 0; i < 8; ++i) t += red[i];
    scale = rsqrtf(t / (float)H + 1e-6f);
  }
  __syncthreads();
  const float sc = scale;
  for (int i = threadIdx.x; i < H; i += 256)
    y[(size_t)row * H + i] = xr[i] * sc * w[i];
}

// Row-wise softmax, in place, fp32.
__global__ __launch_bounds__(256) void softmax_k(float* __restrict__ s, int N)
{
  __shared__ float red[8];
  __shared__ float bval;
  const int row = blockIdx.x;
  float* sr = s + (size_t)row * N;

  float m = -3.4e38f;
  for (int i = threadIdx.x; i < N; i += 256) m = fmaxf(m, sr[i]);
#pragma unroll
  for (int off = 16; off; off >>= 1) m = fmaxf(m, __shfl_down(m, off, 32));
  if ((threadIdx.x & 31) == 0) red[threadIdx.x >> 5] = m;
  __syncthreads();
  if (threadIdx.x == 0) {
    float t = red[0];
    for (int i = 1; i < 8; ++i) t = fmaxf(t, red[i]);
    bval = t;
  }
  __syncthreads();
  const float mx = bval;

  float sum = 0.f;
  for (int i = threadIdx.x; i < N; i += 256) {
    float e = __expf(sr[i] - mx);
    sr[i] = e;
    sum += e;
  }
  __syncthreads();
#pragma unroll
  for (int off = 16; off; off >>= 1) sum += __shfl_down(sum, off, 32);
  if ((threadIdx.x & 31) == 0) red[threadIdx.x >> 5] = sum;
  __syncthreads();
  if (threadIdx.x == 0) {
    float t = 0.f;
    for (int i = 0; i < 8; ++i) t += red[i];
    bval = 1.f / t;
  }
  __syncthreads();
  const float inv = bval;
  for (int i = threadIdx.x; i < N; i += 256) sr[i] *= inv;
}

// ---------------------------------------------------------------------------
template <int EPI, bool BT>
static inline void gemm(hipStream_t st,
                        const float* A, int lda, const float* B, int ldb,
                        float* C, int ldc, const float* R, int ldr,
                        int M, int N, int K, float alpha)
{
  dim3 grid(N / GEMM_BN, M / GEMM_BM), block(256);
  gemm_wmma<EPI, BT><<<grid, block, 0, st>>>(A, lda, B, ldb, C, ldc, R, ldr,
                                             M, N, K, alpha);
}

extern "C" void kernel_launch(void* const* d_in, const int* in_sizes, int n_in,
                              void* d_out, int out_size, void* d_ws, size_t ws_size,
                              hipStream_t stream)
{
  (void)in_sizes; (void)n_in; (void)out_size; (void)ws_size;

  constexpr int S  = 2048;
  constexpr int H  = 1024;
  constexpr int NH = 16;
  constexpr int HD = 64;
  constexpr int I  = 4096;

  const float* x          = (const float*)d_in[0];
  const float* ctx        = (const float*)d_in[1];
  const float* sa_norm_w  = (const float*)d_in[2];
  const float* sa_wq      = (const float*)d_in[3];
  const float* sa_wk      = (const float*)d_in[4];
  const float* sa_wv      = (const float*)d_in[5];
  const float* sa_wo      = (const float*)d_in[6];
  const float* ca_norm_w  = (const float*)d_in[7];
  const float* ca_wq      = (const float*)d_in[8];
  const float* ca_wk      = (const float*)d_in[9];
  const float* ca_wv      = (const float*)d_in[10];
  const float* ca_wo      = (const float*)d_in[11];
  const float* mlp_norm_w = (const float*)d_in[12];
  const float* w_gate     = (const float*)d_in[13];
  const float* w_up       = (const float*)d_in[14];
  const float* w_down     = (const float*)d_in[15];
  float* out = (float*)d_out;

  // workspace layout (fp32 elements), total ~104 MB
  float* ws     = (float*)d_ws;
  float* hn     = ws;                              // S*H
  float* q      = hn     + (size_t)S * H;          // S*H
  float* k      = q      + (size_t)S * H;          // S*H
  float* v      = k      + (size_t)S * H;          // S*H
  float* attn   = v      + (size_t)S * H;          // S*H
  float* h1     = attn   + (size_t)S * H;          // S*H
  float* h2     = h1     + (size_t)S * H;          // S*H
  float* scores = h2     + (size_t)S * H;          // S*S (per-head, reused)
  float* gbuf   = scores + (size_t)S * S;          // S*I

  const float sm_scale = 0.125f;  // HD^-0.5

  // ---------------- self-attention block --------------------------------
  rmsnorm_k<<<S, 256, 0, stream>>>(x, sa_norm_w, hn, H);
  gemm<EPI_NONE, false>(stream, hn, H, sa_wq, H, q, H, nullptr, 0, S, H, H, 1.f);
  gemm<EPI_NONE, false>(stream, hn, H, sa_wk, H, k, H, nullptr, 0, S, H, H, 1.f);
  gemm<EPI_NONE, false>(stream, hn, H, sa_wv, H, v, H, nullptr, 0, S, H, H, 1.f);
  for (int hd = 0; hd < NH; ++hd) {
    gemm<EPI_NONE, true >(stream, q + hd * HD, H, k + hd * HD, H,
                          scores, S, nullptr, 0, S, S, HD, sm_scale);
    softmax_k<<<S, 256, 0, stream>>>(scores, S);
    gemm<EPI_NONE, false>(stream, scores, S, v + hd * HD, H,
                          attn + hd * HD, H, nullptr, 0, S, HD, S, 1.f);
  }
  gemm<EPI_ADD, false>(stream, attn, H, sa_wo, H, h1, H, x, H, S, H, H, 1.f);

  // ---------------- cross-attention block -------------------------------
  rmsnorm_k<<<S, 256, 0, stream>>>(h1, ca_norm_w, hn, H);
  gemm<EPI_NONE, false>(stream, hn,  H, ca_wq, H, q, H, nullptr, 0, S, H, H, 1.f);
  gemm<EPI_NONE, false>(stream, ctx, H, ca_wk, H, k, H, nullptr, 0, S, H, H, 1.f);
  gemm<EPI_NONE, false>(stream, ctx, H, ca_wv, H, v, H, nullptr, 0, S, H, H, 1.f);
  for (int hd = 0; hd < NH; ++hd) {
    gemm<EPI_NONE, true >(stream, q + hd * HD, H, k + hd * HD, H,
                          scores, S, nullptr, 0, S, S, HD, sm_scale);
    softmax_k<<<S, 256, 0, stream>>>(scores, S);
    gemm<EPI_NONE, false>(stream, scores, S, v + hd * HD, H,
                          attn + hd * HD, H, nullptr, 0, S, HD, S, 1.f);
  }
  gemm<EPI_ADD, false>(stream, attn, H, ca_wo, H, h2, H, h1, H, S, H, H, 1.f);

  // ---------------- MLP block -------------------------------------------
  rmsnorm_k<<<S, 256, 0, stream>>>(h2, mlp_norm_w, hn, H);
  // gbuf = silu(hn @ w_gate)
  gemm<EPI_SILU, false>(stream, hn, H, w_gate, I, gbuf, I, nullptr, 0, S, I, H, 1.f);
  // gbuf = (hn @ w_up) * gbuf   (in-place: R index == C index per thread)
  gemm<EPI_MUL,  false>(stream, hn, H, w_up,   I, gbuf, I, gbuf, I, S, I, H, 1.f);
  // out = h2 + gbuf @ w_down
  gemm<EPI_ADD,  false>(stream, gbuf, I, w_down, H, out, H, h2, H, S, H, I, 1.f);
}
